// MPLMSparseDecoderLayer_62294205662049
// MI455X (gfx1250) — compile-verified, matrix operand order checked by reading, not run on hardware
//
#include <hip/hip_runtime.h>

// ---------------- problem constants (from reference) ----------------
#define NN      128
#define QQ      32
#define KKTOK   32
#define E_TOT   1152          // E_R + NN, last NN edges are self loops
#define HIDC    1024
#define HH      8
#define KVHH    2
#define HDD     128
#define KV_TOT  64            // QQ + KKTOK
#define MASKMIN (-3.4028234663852886e+38f)

// ---------------- WMMA types ----------------
typedef __attribute__((ext_vector_type(16))) __bf16 v16bf;
typedef __attribute__((ext_vector_type(8)))  float  v8f;
typedef __attribute__((ext_vector_type(4)))  unsigned int v4u;
typedef __attribute__((ext_vector_type(4)))  int v4i;

__device__ __forceinline__ v16bf load_frag(const __bf16* p0, const __bf16* p1) {
    union { v4u u[2]; v16bf v; } f;
    f.u[0] = *(const v4u*)p0;
    f.u[1] = *(const v4u*)p1;
    return f.v;
}

__device__ __forceinline__ v8f wmma_bf16(v16bf a, v16bf b, v8f c) {
    return __builtin_amdgcn_wmma_f32_16x16x32_bf16(false, a, false, b, (short)0, c,
                                                   false, false);
}

// ---------------- CDNA5 async global->LDS copy (guarded) ----------------
#if defined(__AMDGCN__) && __has_builtin(__builtin_amdgcn_global_load_async_to_lds_b128)
#define HAVE_ASYNC_COPY 1
#else
#define HAVE_ASYNC_COPY 0
#endif

__device__ __forceinline__ void copy16_g2l(const __bf16* g, __bf16* l) {
#if HAVE_ASYNC_COPY
    __builtin_amdgcn_global_load_async_to_lds_b128(
        (__attribute__((address_space(1))) v4i*)(void*)g,
        (__attribute__((address_space(3))) v4i*)(void*)l, 0, 0);
#else
    *(v4u*)l = *(const v4u*)g;   // synchronous fallback through VGPRs
#endif
}

template <int N> __device__ __forceinline__ void wait_async() {
#if HAVE_ASYNC_COPY
#if __has_builtin(__builtin_amdgcn_s_wait_asynccnt)
    __builtin_amdgcn_s_wait_asynccnt((short)N);
#else
    asm volatile("s_wait_asynccnt %0" :: "i"(N) : "memory");
#endif
#endif
}

// ---------------- elementwise kernels ----------------
__global__ void f32_to_bf16_kernel(const float* __restrict__ x,
                                   __bf16* __restrict__ y, size_t n) {
    for (size_t i = (size_t)blockIdx.x * blockDim.x + threadIdx.x; i < n;
         i += (size_t)gridDim.x * blockDim.x)
        y[i] = (__bf16)x[i];
}

// RMSNorm over rows of width 1024, f32 in -> bf16 out. One block per row.
__global__ void rmsnorm_bf16_kernel(const float* __restrict__ x,
                                    const float* __restrict__ w,
                                    __bf16* __restrict__ y) {
    const int row  = blockIdx.x;
    const int tid  = threadIdx.x;
    const int lane = tid & 31, wv = tid >> 5;
    const float* xr = x + (size_t)row * HIDC;
    const int base = tid * 4;
    float x0 = xr[base + 0], x1 = xr[base + 1], x2 = xr[base + 2], x3 = xr[base + 3];
    float s = x0 * x0 + x1 * x1 + x2 * x2 + x3 * x3;
    for (int o = 16; o > 0; o >>= 1) s += __shfl_xor(s, o, 32);
    __shared__ float red[8];
    if (lane == 0) red[wv] = s;
    __syncthreads();
    if (tid == 0) {
        float t = 0.f;
        for (int j = 0; j < 8; ++j) t += red[j];
        red[0] = rsqrtf(t / (float)HIDC + 1e-6f);
    }
    __syncthreads();
    const float sc = red[0];
    __bf16* yr = y + (size_t)row * HIDC;
    yr[base + 0] = (__bf16)(w[base + 0] * x0 * sc);
    yr[base + 1] = (__bf16)(w[base + 1] * x1 * sc);
    yr[base + 2] = (__bf16)(w[base + 2] * x2 * sc);
    yr[base + 3] = (__bf16)(w[base + 3] * x3 * sc);
}

// act = bf16( silu(gate) * up )
__global__ void silu_mul_kernel(const float* __restrict__ g,
                                const float* __restrict__ u,
                                __bf16* __restrict__ out, size_t n) {
    for (size_t i = (size_t)blockIdx.x * blockDim.x + threadIdx.x; i < n;
         i += (size_t)gridDim.x * blockDim.x) {
        float x = g[i];
        float s = x / (1.f + __expf(-x));
        out[i] = (__bf16)(s * u[i]);
    }
}

// ---------------- LDS-staged bf16 WMMA GEMM: C[M,N] = A[M,K] * W[N,K]^T ----------------
// Block 256 threads (8 waves). Block tile 128x64, wave (wr,wc) owns 32x32.
// Double-buffered LDS tiles filled with async global->LDS copies (ASYNCcnt).
#define TM 128
#define TN 64
#define TK 32

template <bool OUT_BF16>
__global__ __launch_bounds__(256) void gemm_bf16_kernel(
    const __bf16* __restrict__ A, const __bf16* __restrict__ W,
    float* __restrict__ Cf, __bf16* __restrict__ Cb,
    const float* __restrict__ resid, int M, int Nn, int Kk) {
    __shared__ __align__(16) __bf16 As[2][TM][TK];
    __shared__ __align__(16) __bf16 Bs[2][TN][TK];

    const int tid  = threadIdx.x;
    const int lane = tid & 31, w = tid >> 5;
    const int wr = w >> 1, wc = w & 1;
    const int r  = lane & 15, kh = lane >> 4;
    const int m0 = blockIdx.x * TM;
    const int n0 = blockIdx.y * TN;

    // staging assignment: 16B chunks; A has 512 chunks (2/thread), B has 256 (1/thread)
    const int ar0 = (tid * 2) >> 2,     ac0 = ((tid * 2) & 3) * 8;
    const int ar1 = (tid * 2 + 1) >> 2, ac1 = ((tid * 2 + 1) & 3) * 8;
    const int br  = tid >> 2,           bc  = (tid & 3) * 8;

    const __bf16* Ag = A + (size_t)m0 * Kk;
    const __bf16* Wg = W + (size_t)n0 * Kk;

#define STAGE(buf, k0)                                                     \
    do {                                                                   \
        copy16_g2l(Ag + (size_t)ar0 * Kk + (k0) + ac0, &As[buf][ar0][ac0]);\
        copy16_g2l(Ag + (size_t)ar1 * Kk + (k0) + ac1, &As[buf][ar1][ac1]);\
        copy16_g2l(Wg + (size_t)br  * Kk + (k0) + bc,  &Bs[buf][br][bc]);  \
    } while (0)

    v8f acc00 = {}, acc01 = {}, acc10 = {}, acc11 = {};
    const int nk = Kk / TK;
    STAGE(0, 0);
    for (int s = 0; s < nk; ++s) {
        const int buf = s & 1;
        if (s + 1 < nk) {
            STAGE(buf ^ 1, (s + 1) * TK);
            wait_async<3>();     // previous stage's 3 copies complete (in-order)
        } else {
            wait_async<0>();
        }
        __syncthreads();
        const __bf16* a0 = &As[buf][wr * 32 + r][kh * 8];
        const __bf16* a1 = &As[buf][wr * 32 + 16 + r][kh * 8];
        const __bf16* b0 = &Bs[buf][wc * 32 + r][kh * 8];
        const __bf16* b1 = &Bs[buf][wc * 32 + 16 + r][kh * 8];
        v16bf af0 = load_frag(a0, a0 + 16);
        v16bf af1 = load_frag(a1, a1 + 16);
        v16bf bf0 = load_frag(b0, b0 + 16);
        v16bf bf1 = load_frag(b1, b1 + 16);
        acc00 = wmma_bf16(af0, bf0, acc00);
        acc01 = wmma_bf16(af0, bf1, acc01);
        acc10 = wmma_bf16(af1, bf0, acc10);
        acc11 = wmma_bf16(af1, bf1, acc11);
        __syncthreads();         // all waves done reading buf before it is refilled
    }
#undef STAGE

    for (int g = 0; g < 8; ++g) {
        const int mA = m0 + wr * 32 + g + kh * 8;       // row tile 0
        const int mB = mA + 16;                          // row tile 1
        const int nA = n0 + wc * 32 + r;                 // col tile 0 (+16 => tile 1)
        const size_t iA = (size_t)mA * Nn + nA;
        const size_t iB = (size_t)mB * Nn + nA;
        float v00 = acc00[g], v01 = acc01[g], v10 = acc10[g], v11 = acc11[g];
        if (resid) {
            v00 += resid[iA]; v01 += resid[iA + 16];
            v10 += resid[iB]; v11 += resid[iB + 16];
        }
        if (OUT_BF16) {
            Cb[iA] = (__bf16)v00; Cb[iA + 16] = (__bf16)v01;
            Cb[iB] = (__bf16)v10; Cb[iB + 16] = (__bf16)v11;
        } else {
            Cf[iA] = v00; Cf[iA + 16] = v01;
            Cf[iB] = v10; Cf[iB + 16] = v11;
        }
    }
}

// ---------------- query RoPE: q f32 [N*Q, H*HD] -> bf16 [N,H,Q,HD] ----------------
__global__ void rope_q_kernel(const float* __restrict__ q,
                              const int* __restrict__ node_order,
                              __bf16* __restrict__ out) {
    const int b = blockIdx.x;          // n*32 + t
    const int n = b >> 5, t = b & 31;
    const float pos = (float)(node_order[n] + t);
    for (int i = threadIdx.x; i < HH * HDD; i += 256) {
        const int h = i >> 7, d = i & 127;
        const int dh = d & 63;
        const float inv = __powf(10000.f, -(float)(2 * dh) / 128.f);
        const float ang = pos * inv;
        const float cs = __cosf(ang), sn = __sinf(ang);
        const float x  = q[(size_t)b * HIDC + i];
        const float xo = q[(size_t)b * HIDC + h * 128 + (d ^ 64)];
        const float r = (d < 64) ? (x * cs - xo * sn) : (x * cs + xo * sn);
        out[(size_t)(((n * HH + h) * QQ) + t) * HDD + d] = (__bf16)r;
    }
}

// ---------------- per-edge masked scores via WMMA ----------------
__global__ __launch_bounds__(256) void scores_kernel(
    const __bf16* __restrict__ qb,   // [N,H,32,128] roped bf16
    const float* __restrict__ kf,    // [N*32, 256] f32 (pre-rope)
    const float* __restrict__ ekf,   // [E*32, 256] f32 (pre-rope)
    const int* __restrict__ src_arr, const int* __restrict__ dst_arr,
    const int* __restrict__ mem_mask, const int* __restrict__ edge_mask,
    const int* __restrict__ edge_order,
    float* __restrict__ scores) {    // [E,8,32,64]
    const int e   = blockIdx.x;
    const int src = src_arr[e], dst = dst_arr[e];
    const bool self_e = (e >= E_TOT - NN);

    __shared__ int   om[KV_TOT];
    __shared__ int   kpos[KV_TOT];
    __shared__ float colmask[KV_TOT];
    __shared__ __align__(16) __bf16 Kt[KVHH][KV_TOT][HDD];

    for (int j = threadIdx.x; j < KV_TOT; j += 256) {
        const int m = (j < QQ) ? mem_mask[src * QQ + j] : edge_mask[e * KKTOK + (j - QQ)];
        om[j] = m;
        const int meff = (self_e && j >= QQ) ? 0 : m;
        colmask[j] = (1.f - (float)meff) * MASKMIN;
    }
    __syncthreads();
    if (threadIdx.x == 0) {
        int acc = edge_order[e];
        kpos[0] = acc;
        for (int j = 1; j < KV_TOT; ++j) { acc += om[j]; kpos[j] = acc; }
    }
    __syncthreads();

    // RoPE the 64 K rows (32 node-K + 32 edge-K) for both KV heads into LDS bf16
    for (int i = threadIdx.x; i < KVHH * KV_TOT * 64; i += 256) {
        const int c  = i >> 12;          // kv head
        const int rr = (i & 4095) >> 6;  // kv row 0..63
        const int dh = i & 63;
        const float pos = (float)kpos[rr];
        const float inv = __powf(10000.f, -(float)(2 * dh) / 128.f);
        const float ang = pos * inv;
        const float cs = __cosf(ang), sn = __sinf(ang);
        const float* sp = (rr < QQ)
            ? kf  + ((size_t)(src * QQ + rr)) * (KVHH * HDD) + c * HDD
            : ekf + ((size_t)(e * KKTOK + (rr - QQ))) * (KVHH * HDD) + c * HDD;
        const float x1 = sp[dh], x2 = sp[dh + 64];
        Kt[c][rr][dh]      = (__bf16)(x1 * cs - x2 * sn);
        Kt[c][rr][dh + 64] = (__bf16)(x2 * cs + x1 * sn);
    }
    __syncthreads();

    const int lane = threadIdx.x & 31;
    const int w    = threadIdx.x >> 5;       // wave = head
    const int r  = lane & 15;
    const int kh = lane >> 4;
    const float scale = 0.08838834764831845f;  // 1/sqrt(128)
    for (int tloc = 0; tloc < 8; ++tloc) {
        const int h  = w;
        const int qt = (tloc >> 2) & 1;
        const int kt = tloc & 3;
        const int c  = h >> 2;
        const int q0 = qt * 16, kv0 = kt * 16;
        v8f acc = {};
        const __bf16* ap = qb + (size_t)((dst * HH + h) * QQ + q0 + r) * HDD + kh * 8;
        const __bf16* bp = &Kt[c][kv0 + r][kh * 8];
        for (int k0 = 0; k0 < HDD; k0 += 32) {
            v16bf af  = load_frag(ap + k0, ap + k0 + 16);
            v16bf bf_ = load_frag(bp + k0, bp + k0 + 16);
            acc = wmma_bf16(af, bf_, acc);
        }
        const int kv = kv0 + r;
        const float cm = colmask[kv];
        for (int g = 0; g < 8; ++g) {
            const int q = q0 + g + kh * 8;
            float add = cm;
            if (self_e && kv < QQ) add += (kv <= q) ? 0.f : MASKMIN;
            scores[(((size_t)e * HH + h) * QQ + q) * KV_TOT + kv] = acc[g] * scale + add;
        }
    }
}

// ---------------- deterministic CSR of edges grouped by dst ----------------
__global__ void build_csr_kernel(const int* __restrict__ dst_arr,
                                 int* __restrict__ off, int* __restrict__ elist) {
    if (threadIdx.x == 0 && blockIdx.x == 0) {
        int cnt[NN];
        for (int n = 0; n < NN; ++n) cnt[n] = 0;
        for (int e = 0; e < E_TOT; ++e) cnt[dst_arr[e]]++;
        int acc = 0;
        for (int n = 0; n < NN; ++n) { off[n] = acc; acc += cnt[n]; cnt[n] = off[n]; }
        off[NN] = acc;
        for (int e = 0; e < E_TOT; ++e) elist[cnt[dst_arr[e]]++] = e;  // stable order
    }
}

// ---------------- segment softmax + P*V via WMMA; one block per (dst,head) ----------------
__global__ __launch_bounds__(256) void attn_av_kernel(
    const float* __restrict__ scores,   // [E,8,32,64]
    const __bf16* __restrict__ vb,      // [N*32, 256]
    const __bf16* __restrict__ evb,     // [E*32, 256]
    const int* __restrict__ src_arr,
    const int* __restrict__ off, const int* __restrict__ elist,
    __bf16* __restrict__ attn) {        // [N*32, 1024] columns h*128+d
    const int n = blockIdx.x >> 3;
    const int h = blockIdx.x & 7;
    const int c = h >> 2;
    const int beg = off[n], end = off[n + 1];
    const int tid = threadIdx.x;
    const int q = tid & 31, chunk = tid >> 5;   // (q, 8-col chunk)

    __shared__ float red[32][8];
    __shared__ float mx[32], den[32];
    __shared__ __align__(16) __bf16 Pt[QQ][KV_TOT];
    __shared__ __align__(16) __bf16 VT[HDD][KV_TOT];

    // pass 1: segment max per query row
    float lm = MASKMIN;
    for (int ei = beg; ei < end; ++ei) {
        const int e = elist[ei];
        const float* sp = scores + (((size_t)e * HH + h) * QQ + q) * KV_TOT + chunk * 8;
        if (ei + 1 < end) {
            const int en = elist[ei + 1];
            __builtin_prefetch(scores + (((size_t)en * HH + h) * QQ + q) * KV_TOT, 0, 1);
        }
        for (int j = 0; j < 8; ++j) lm = fmaxf(lm, sp[j]);
    }
    red[q][chunk] = lm;
    __syncthreads();
    if (tid < 32) {
        float m = red[tid][0];
        for (int j = 1; j < 8; ++j) m = fmaxf(m, red[tid][j]);
        mx[tid] = m;
    }
    __syncthreads();
    const float myMax = mx[q];

    // pass 2: accumulate exp and P*V
    const int lane = tid & 31, w = tid >> 5;
    const int r = lane & 15, kh = lane >> 4;
    const int q0 = (w >> 2) * 16;
    const int d0 = (w & 3) * 32;
    v8f acc0 = {}, acc1 = {};
    float dl = 0.f;

    for (int ei = beg; ei < end; ++ei) {
        const int e = elist[ei];
        const int s = src_arr[e];
        const float* sp = scores + (((size_t)e * HH + h) * QQ + q) * KV_TOT + chunk * 8;
        for (int j = 0; j < 8; ++j) {
            const float p = __expf(sp[j] - myMax);
            dl += p;
            Pt[q][chunk * 8 + j] = (__bf16)p;
        }
        for (int i = tid; i < KV_TOT * HDD; i += 256) {
            const int rr = i >> 7, d = i & 127;
            const __bf16 val = (rr < QQ)
                ? vb[((size_t)(s * QQ + rr)) * (KVHH * HDD) + c * HDD + d]
                : evb[((size_t)(e * KKTOK + (rr - QQ))) * (KVHH * HDD) + c * HDD + d];
            VT[d][rr] = val;
        }
        __syncthreads();
        const __bf16* apw = &Pt[q0 + r][kh * 8];
        const __bf16* bp0 = &VT[d0 + r][kh * 8];
        const __bf16* bp1 = &VT[d0 + 16 + r][kh * 8];
        for (int k0 = 0; k0 < KV_TOT; k0 += 32) {
            v16bf af = load_frag(apw + k0, apw + k0 + 16);
            acc0 = wmma_bf16(af, load_frag(bp0 + k0, bp0 + k0 + 16), acc0);
            acc1 = wmma_bf16(af, load_frag(bp1 + k0, bp1 + k0 + 16), acc1);
        }
        __syncthreads();
    }

    red[q][chunk] = dl;
    __syncthreads();
    if (tid < 32) {
        float sden = 0.f;
        for (int j = 0; j < 8; ++j) sden += red[tid][j];
        den[tid] = sden + 1e-16f;
    }
    __syncthreads();
    for (int g = 0; g < 8; ++g) {
        const int qq = q0 + g + kh * 8;
        const float inv = 1.f / den[qq];
        const size_t o = ((size_t)(n * QQ + qq)) * HIDC + h * HDD + d0 + r;
        attn[o]      = (__bf16)(acc0[g] * inv);
        attn[o + 16] = (__bf16)(acc1[g] * inv);
    }
}

// ---------------- host launcher ----------------
extern "C" void kernel_launch(void* const* d_in, const int* in_sizes, int n_in,
                              void* d_out, int out_size, void* d_ws, size_t ws_size,
                              hipStream_t stream) {
    (void)in_sizes; (void)n_in; (void)out_size;
    const float* hidden     = (const float*)d_in[0];
    const float* edgeh      = (const float*)d_in[1];
    const int*   eidx       = (const int*)d_in[2];
    const int*   mem_mask   = (const int*)d_in[3];
    const int*   edge_mask  = (const int*)d_in[4];
    const int*   node_order = (const int*)d_in[5];
    const int*   edge_order = (const int*)d_in[6];
    const float* Wq    = (const float*)d_in[8];
    const float* Wk    = (const float*)d_in[9];
    const float* Wv    = (const float*)d_in[10];
    const float* Wo    = (const float*)d_in[11];
    const float* ln1   = (const float*)d_in[12];
    const float* ln2   = (const float*)d_in[13];
    const float* Wgate = (const float*)d_in[14];
    const float* Wup   = (const float*)d_in[15];
    const float* Wdown = (const float*)d_in[16];
    const int* src_arr = eidx;
    const int* dst_arr = eidx + E_TOT;

    char* ws = (char*)d_ws;
    size_t cur = 0;
    auto arena = [&](size_t bytes) -> char* {
        char* p = ws + cur;
        cur = (cur + bytes + 255) & ~(size_t)255;
        return p;
    };
    __bf16* hN   = (__bf16*)arena((size_t)4096 * 1024 * 2);
    __bf16* eN   = (__bf16*)arena((size_t)36864 * 1024 * 2);   // aliased later by gate f32
    __bf16* Wq_b = (__bf16*)arena((size_t)1024 * 1024 * 2);
    __bf16* Wk_b = (__bf16*)arena((size_t)256 * 1024 * 2);
    __bf16* Wv_b = (__bf16*)arena((size_t)256 * 1024 * 2);
    __bf16* Wo_b = (__bf16*)arena((size_t)1024 * 1024 * 2);
    __bf16* Wg_b = (__bf16*)arena((size_t)4096 * 1024 * 2);
    __bf16* Wu_b = (__bf16*)arena((size_t)4096 * 1024 * 2);
    __bf16* Wd_b = (__bf16*)arena((size_t)1024 * 4096 * 2);
    float*  qf   = (float*)arena((size_t)4096 * 1024 * 4);
    __bf16* qb   = (__bf16*)arena((size_t)4096 * 1024 * 2);
    float*  kf   = (float*)arena((size_t)4096 * 256 * 4);
    float*  ekf  = (float*)arena((size_t)36864 * 256 * 4);
    __bf16* vb   = (__bf16*)arena((size_t)4096 * 256 * 2);
    __bf16* evb  = (__bf16*)arena((size_t)36864 * 256 * 2);
    float*  sc   = (float*)arena((size_t)E_TOT * 8 * 32 * 64 * 4); // aliased later by up f32
    int*    csr_off  = (int*)arena((size_t)(NN + 1) * 4);
    int*    csr_list = (int*)arena((size_t)E_TOT * 4);
    __bf16* attnb = (__bf16*)arena((size_t)4096 * 1024 * 2);
    float*  h2    = (float*)arena((size_t)4096 * 1024 * 4);
    __bf16* h2n   = (__bf16*)arena((size_t)4096 * 1024 * 2);
    __bf16* act   = (__bf16*)arena((size_t)4096 * 4096 * 2);
    float* gate = (float*)eN;   // 67.1MB <= 75.5MB region, eN dead by then
    float* up   = (float*)sc;   // 67.1MB <= 75.5MB region, scores dead by then
    if (cur > ws_size) return;  // insufficient workspace; nothing safe to do

    const dim3 blk(256);
    // 1) weights -> bf16
    f32_to_bf16_kernel<<<512, blk, 0, stream>>>(Wq, Wq_b, (size_t)1024 * 1024);
    f32_to_bf16_kernel<<<512, blk, 0, stream>>>(Wk, Wk_b, (size_t)256 * 1024);
    f32_to_bf16_kernel<<<512, blk, 0, stream>>>(Wv, Wv_b, (size_t)256 * 1024);
    f32_to_bf16_kernel<<<512, blk, 0, stream>>>(Wo, Wo_b, (size_t)1024 * 1024);
    f32_to_bf16_kernel<<<512, blk, 0, stream>>>(Wgate, Wg_b, (size_t)4096 * 1024);
    f32_to_bf16_kernel<<<512, blk, 0, stream>>>(Wup, Wu_b, (size_t)4096 * 1024);
    f32_to_bf16_kernel<<<512, blk, 0, stream>>>(Wdown, Wd_b, (size_t)1024 * 4096);
    // 2) RMSNorm
    rmsnorm_bf16_kernel<<<4096, blk, 0, stream>>>(hidden, ln1, hN);
    rmsnorm_bf16_kernel<<<36864, blk, 0, stream>>>(edgeh, ln1, eN);
    // 3) QKV projections
    gemm_bf16_kernel<false><<<dim3(32, 16), blk, 0, stream>>>(hN, Wq_b, qf, nullptr, nullptr, 4096, 1024, 1024);
    gemm_bf16_kernel<false><<<dim3(32, 4),  blk, 0, stream>>>(hN, Wk_b, kf, nullptr, nullptr, 4096, 256, 1024);
    gemm_bf16_kernel<true ><<<dim3(32, 4),  blk, 0, stream>>>(hN, Wv_b, nullptr, vb, nullptr, 4096, 256, 1024);
    gemm_bf16_kernel<false><<<dim3(288, 4), blk, 0, stream>>>(eN, Wk_b, ekf, nullptr, nullptr, 36864, 256, 1024);
    gemm_bf16_kernel<true ><<<dim3(288, 4), blk, 0, stream>>>(eN, Wv_b, nullptr, evb, nullptr, 36864, 256, 1024);
    // 4) query RoPE, CSR, scores, segment softmax + AV
    rope_q_kernel<<<4096, blk, 0, stream>>>(qf, node_order, qb);
    build_csr_kernel<<<1, 32, 0, stream>>>(dst_arr, csr_off, csr_list);
    scores_kernel<<<E_TOT, blk, 0, stream>>>(qb, kf, ekf, src_arr, dst_arr,
                                             mem_mask, edge_mask, edge_order, sc);
    attn_av_kernel<<<NN * HH, blk, 0, stream>>>(sc, vb, evb, src_arr,
                                                csr_off, csr_list, attnb);
    // 5) output projection + residual -> h2; norm -> h2n
    gemm_bf16_kernel<false><<<dim3(32, 16), blk, 0, stream>>>(attnb, Wo_b, h2, nullptr, hidden, 4096, 1024, 1024);
    rmsnorm_bf16_kernel<<<4096, blk, 0, stream>>>(h2, ln2, h2n);
    // 6) MLP
    gemm_bf16_kernel<false><<<dim3(32, 64), blk, 0, stream>>>(h2n, Wg_b, gate, nullptr, nullptr, 4096, 4096, 1024);
    gemm_bf16_kernel<false><<<dim3(32, 64), blk, 0, stream>>>(h2n, Wu_b, up, nullptr, nullptr, 4096, 4096, 1024);
    silu_mul_kernel<<<8192, blk, 0, stream>>>(gate, up, act, (size_t)4096 * 4096);
    gemm_bf16_kernel<false><<<dim3(32, 16), blk, 0, stream>>>(act, Wd_b, (float*)d_out, nullptr, h2, 4096, 1024, 4096);
}
